// MockGCNModel_69286412419644
// MI455X (gfx1250) — matrix operand order repaired and checked by reading, not compiled, not run on hardware
//
#include <hip/hip_runtime.h>
#include <math.h>

typedef __attribute__((ext_vector_type(16))) _Float16 v16h;
typedef __attribute__((ext_vector_type(8)))  _Float16 v8h;
typedef __attribute__((ext_vector_type(8)))  float    v8f;

#define F_DIM 32
#define H_DIM 64
#define WAVES_PER_BLOCK 8
#define BLOCK_THREADS 256

// single-instruction ReLU: v_med3_f32(x, 0, +inf)
__device__ __forceinline__ float relu1(float x) {
    return __builtin_amdgcn_fmed3f(x, 0.0f, __builtin_inff());
}

// ---- fragment loaders ------------------------------------------------------
// B-matrix (KxN = (kOff..kOff+32) x 16 cols) fragment, f16, per ISA layout:
//   lanes 0-15: col = lane,  K = kOff+0..7 and kOff+16..23
//   lanes16-31: col = lane-16, K = kOff+8..15 and kOff+24..31
__device__ __forceinline__ v16h load_b_frag(const float* __restrict__ W,
                                            int kOff, int nOff, int lh, int hf) {
    v16h f;
#pragma unroll
    for (int j = 0; j < 16; ++j) {
        int k = kOff + 8 * hf + (j < 8 ? j : j + 8);
        f[j] = (_Float16)W[k * H_DIM + nOff + lh];
    }
    return f;
}

// A-matrix 16x32 f16 fragment from row-major x[row][0..31]:
//   lanes 0-15: row = lane,  K = 0..7 and 16..23
//   lanes16-31: row = lane-16, K = 8..15 and 24..31
__device__ __forceinline__ v16h load_a_x(const float* __restrict__ x,
                                         long long row, int hf) {
    const float* p = x + row * F_DIM + 8 * hf;
    float4 f0 = *(const float4*)(p + 0);
    float4 f1 = *(const float4*)(p + 4);
    float4 f2 = *(const float4*)(p + 16);
    float4 f3 = *(const float4*)(p + 20);
    float v[16] = {f0.x, f0.y, f0.z, f0.w, f1.x, f1.y, f1.z, f1.w,
                   f2.x, f2.y, f2.z, f2.w, f3.x, f3.y, f3.z, f3.w};
    v16h a;
#pragma unroll
    for (int j = 0; j < 16; ++j) a[j] = (_Float16)v[j];
    return a;
}

// ---- fused 2-layer MLP + segment accumulation ------------------------------
__global__ void __launch_bounds__(BLOCK_THREADS)
gcn_fused_kernel(const float* __restrict__ x, const int* __restrict__ batch,
                 const float* __restrict__ W_in, const float* __restrict__ b_in,
                 const float* __restrict__ W_h,  const float* __restrict__ b_h,
                 float* __restrict__ seg_sum, float* __restrict__ counts,
                 long long numTiles) {
    __shared__ __align__(16) _Float16 hbuf[WAVES_PER_BLOCK * 16 * H_DIM];

    const int lane  = threadIdx.x & 31;
    const int lh    = lane & 15;   // row/col within 16
    const int hf    = lane >> 4;   // which K/M half this lane carries
    const int wslot = threadIdx.x >> 5;
    _Float16* hw = hbuf + wslot * (16 * H_DIM);

    // Preload weight B-fragments + bias-filled C fragments once per wave.
    v16h bIn[4];        // W_in: 32x64 -> 4 col tiles of 32x16
    v16h bH[4][2];      // W_h : 64x64 -> 4 col tiles x 2 K-halves of 32x16
    v8f  cbIn[4], cbH[4];   // C fragments pre-loaded with bias (col-only value)
#pragma unroll
    for (int t = 0; t < 4; ++t) {
        bIn[t]   = load_b_frag(W_in, 0,  16 * t, lh, hf);
        bH[t][0] = load_b_frag(W_h,  0,  16 * t, lh, hf);
        bH[t][1] = load_b_frag(W_h,  32, 16 * t, lh, hf);
        float bi = b_in[16 * t + lh];
        float bh = b_h[16 * t + lh];
#pragma unroll
        for (int e = 0; e < 8; ++e) { cbIn[t][e] = bi; cbH[t][e] = bh; }
    }

    const long long wave   = (long long)blockIdx.x * WAVES_PER_BLOCK + wslot;
    const long long stride = (long long)gridDim.x * WAVES_PER_BLOCK;

    for (long long tile = wave; tile < numTiles; tile += stride) {
        const long long base = tile * 16;

        // prefetch the next tile this wave will touch (speculative, GL2)
        long long nbase = (tile + stride) * 16;
        if (nbase < numTiles * 16)
            __builtin_prefetch(x + nbase * F_DIM + lane * 16, 0, 1);

        // ---- layer 1: h1 = relu(x_tile @ W_in + b_in) ----
        v16h a1 = load_a_x(x, base + lh, hf);
        v8f c1[4];
#pragma unroll
        for (int t = 0; t < 4; ++t) {
            v8f c = __builtin_amdgcn_wmma_f32_16x16x32_f16(false, a1, false, bIn[t],
                                                           (short)0, cbIn[t],
                                                           false, false);
#pragma unroll
            for (int e = 0; e < 8; ++e) c[e] = relu1(c[e]);
            c1[t] = c;
        }

        // ---- relayout h1 (C-frag) -> A-frag via per-wave LDS tile ----
        // C layout: element r of lane = h1[row r + 8*hf][16*t + lh]
#pragma unroll
        for (int t = 0; t < 4; ++t)
#pragma unroll
            for (int r = 0; r < 8; ++r)
                hw[(r + 8 * hf) * H_DIM + 16 * t + lh] = (_Float16)c1[t][r];
        // same-wave LDS ordering via DScnt; no workgroup barrier needed

        v16h a2[2];
#pragma unroll
        for (int kh = 0; kh < 2; ++kh) {
            const _Float16* rp = hw + lh * H_DIM + 32 * kh + 8 * hf;
            v8h lo = *(const v8h*)(rp);       // K = 8*hf+0..7   (+32*kh)
            v8h hi = *(const v8h*)(rp + 16);  // K = 8*hf+16..23 (+32*kh)
            v16h a;
#pragma unroll
            for (int j = 0; j < 8; ++j) { a[j] = lo[j]; a[j + 8] = hi[j]; }
            a2[kh] = a;
        }

        // ---- layer 2: h2 = relu(h1 @ W_h + b_h), K=64 as two K=32 WMMAs ----
        v8f c2[4];
#pragma unroll
        for (int t = 0; t < 4; ++t) {
            v8f c = __builtin_amdgcn_wmma_f32_16x16x32_f16(false, a2[0], false, bH[t][0],
                                                           (short)0, cbH[t],
                                                           false, false);
            c = __builtin_amdgcn_wmma_f32_16x16x32_f16(false, a2[1], false, bH[t][1],
                                                       (short)0, c, false, false);
#pragma unroll
            for (int e = 0; e < 8; ++e) c[e] = relu1(c[e]);
            c2[t] = c;
        }

        // ---- segment accumulation (batch is sorted) ----
        int g0 = batch[base];
        int g1 = batch[base + 15];
        if (g0 == g1) {
            // whole tile in one graph: reduce 16 rows in-fragment, 1 atomic/lane
#pragma unroll
            for (int t = 0; t < 4; ++t) {
                float s = 0.0f;
#pragma unroll
                for (int r = 0; r < 8; ++r) s += c2[t][r];
                s += __shfl_xor(s, 16, 32);   // combine the two row-halves
                if (lane < 16)
                    atomicAdd(&seg_sum[(size_t)g0 * H_DIM + 16 * t + lh], s);
            }
            if (lane == 0) atomicAdd(&counts[g0], 16.0f);
        } else {
            // graph boundary inside tile (rare): per-row atomics
#pragma unroll
            for (int r = 0; r < 8; ++r) {
                long long node = base + r + 8 * hf;
                int g = batch[node];
#pragma unroll
                for (int t = 0; t < 4; ++t)
                    atomicAdd(&seg_sum[(size_t)g * H_DIM + 16 * t + lh], c2[t][r]);
                if (lh == 0) atomicAdd(&counts[g], 1.0f);
            }
        }
    }
}

// ---- scalar tail for N % 16 leftover nodes (none for N = 2e6) --------------
__global__ void gcn_tail_kernel(const float* __restrict__ x,
                                const int* __restrict__ batch,
                                const float* __restrict__ W_in, const float* __restrict__ b_in,
                                const float* __restrict__ W_h,  const float* __restrict__ b_h,
                                float* __restrict__ seg_sum, float* __restrict__ counts,
                                long long start, int count) {
    int i = blockIdx.x * blockDim.x + threadIdx.x;
    if (i >= count) return;
    long long node = start + i;
    float h1[H_DIM], h2[H_DIM];
    for (int c = 0; c < H_DIM; ++c) {
        float a = b_in[c];
        for (int k = 0; k < F_DIM; ++k) a += x[node * F_DIM + k] * W_in[k * H_DIM + c];
        h1[c] = relu1(a);
    }
    for (int c = 0; c < H_DIM; ++c) {
        float a = b_h[c];
        for (int k = 0; k < H_DIM; ++k) a += h1[k] * W_h[k * H_DIM + c];
        h2[c] = relu1(a);
    }
    int g = batch[node];
    for (int c = 0; c < H_DIM; ++c) atomicAdd(&seg_sum[(size_t)g * H_DIM + c], h2[c]);
    atomicAdd(&counts[g], 1.0f);
}

// ---- mean pooling + output projection --------------------------------------
__global__ void finalize_kernel(const float* __restrict__ seg_sum,
                                const float* __restrict__ counts,
                                const float* __restrict__ W_out,
                                const float* __restrict__ b_out,
                                float* __restrict__ out, int G) {
    int g = blockIdx.x * blockDim.x + threadIdx.x;
    if (g >= G) return;
    float c   = counts[g];
    float inv = (c > 0.0f) ? (1.0f / c) : 0.0f;   // empty graph -> mean = 0
    float acc[5];
#pragma unroll
    for (int j = 0; j < 5; ++j) acc[j] = b_out[j];
    for (int k = 0; k < H_DIM; ++k) {
        float m = seg_sum[(size_t)g * H_DIM + k] * inv;
#pragma unroll
        for (int j = 0; j < 5; ++j) acc[j] += m * W_out[k * 5 + j];
    }
#pragma unroll
    for (int j = 0; j < 5; ++j) out[g * 5 + j] = acc[j];
}

extern "C" void kernel_launch(void* const* d_in, const int* in_sizes, int n_in,
                              void* d_out, int out_size, void* d_ws, size_t ws_size,
                              hipStream_t stream) {
    const float* x     = (const float*)d_in[0];
    const int*   batch = (const int*)d_in[1];   // JAX default config: int32
    const float* W_in  = (const float*)d_in[3];
    const float* b_in  = (const float*)d_in[4];
    const float* W_h   = (const float*)d_in[5];
    const float* b_h   = (const float*)d_in[6];
    const float* W_out = (const float*)d_in[7];
    const float* b_out = (const float*)d_in[8];
    float* out = (float*)d_out;

    long long N = (long long)in_sizes[0] / F_DIM;
    int G = out_size / 5;

    float* seg = (float*)d_ws;
    float* cnt = seg + (size_t)G * H_DIM;
    hipMemsetAsync(d_ws, 0, ((size_t)G * H_DIM + (size_t)G) * sizeof(float), stream);

    long long numTiles = N / 16;
    int tail = (int)(N % 16);

    if (numTiles > 0) {
        int blocks = 1024;   // persistent waves: 8192 waves, ~15 tiles each
        gcn_fused_kernel<<<blocks, BLOCK_THREADS, 0, stream>>>(
            x, batch, W_in, b_in, W_h, b_h, seg, cnt, numTiles);
    }
    if (tail > 0) {
        gcn_tail_kernel<<<1, tail, 0, stream>>>(
            x, batch, W_in, b_in, W_h, b_h, seg, cnt, numTiles * 16, tail);
    }
    finalize_kernel<<<(G + 255) / 256, 256, 0, stream>>>(seg, cnt, W_out, b_out, out, G);
}
